// GNNEmbedder_67405216744186
// MI455X (gfx1250) — matrix operand order repaired and checked by reading, not compiled, not run on hardware
//
#include <hip/hip_runtime.h>

#define NNODE 50000
#define NEDGE 800000
#define DIN_  64
#define D_    128

typedef _Float16 half8_t __attribute__((ext_vector_type(8)));
typedef _Float16 v16h    __attribute__((ext_vector_type(16)));
typedef float    v8f     __attribute__((ext_vector_type(8)));

// ---------------- utility kernels ----------------

__global__ void k_zero_f32(float* __restrict__ p, int n4) {
  int i = blockIdx.x * blockDim.x + threadIdx.x;
  if (i < n4) ((float4*)p)[i] = make_float4(0.f, 0.f, 0.f, 0.f);
}

__global__ void k_f32_to_f16(const float* __restrict__ in, _Float16* __restrict__ out, int n) {
  int i = blockIdx.x * blockDim.x + threadIdx.x;
  if (i < n) out[i] = (_Float16)in[i];
}

// Pack row-major W[K,Kout] (f32) into WMMA B-operand tile layout (f16):
// per (ktile=32, ntile=16) tile: 32 lanes x 16 contiguous halves.
// lanes 0-15 hold k%32 in [0,16): half j -> (k=kt*32+j,    n=nt*16+lane)
// lanes 16-31 hold k%32 in [16,32): half j -> (k=kt*32+16+j, n=nt*16+lane-16)
__global__ void k_pack_b(const float* __restrict__ W, _Float16* __restrict__ Bp, int K, int Kout) {
  int idx = blockIdx.x * blockDim.x + threadIdx.x;
  if (idx >= K * Kout) return;
  int k = idx / Kout, n = idx % Kout;
  int kt = k >> 5, kk = k & 31, nt = n >> 4, nn = n & 15;
  int lane = (kk < 16) ? nn : (16 + nn);
  int j = kk & 15;
  int nNt = Kout >> 4;
  Bp[(((size_t)(kt * nNt + nt)) * 32 + lane) * 16 + j] = (_Float16)W[idx];
}

__global__ void k_count(const int* __restrict__ col, float* __restrict__ cnt, int E) {
  int e = blockIdx.x * blockDim.x + threadIdx.x;
  if (e < E) atomicAdd(&cnt[col[e]], 1.0f);
}

// ---------------- WMMA GEMM: C[M,Kout] = act(A[M,K] @ W + bias) ----------------
// One wave computes a 16x64 C strip (4 N-tiles): A loaded once per k-step feeds
// 4 independent back-to-back v_wmma_f32_16x16x32_f16 (4x A-reuse, deep XDL chain).
__global__ __launch_bounds__(256)
void k_gemm_wmma(const _Float16* __restrict__ A, const _Float16* __restrict__ Bp,
                 const float* __restrict__ bias, float* __restrict__ outF,
                 _Float16* __restrict__ outH, int K, int Kout, int leaky, int nWaves)
{
  const int lane = threadIdx.x & 31;
  const int wave = blockIdx.x * (blockDim.x >> 5) + (threadIdx.x >> 5);
  if (wave >= nWaves) return;            // wave-uniform: EXEC stays all-ones for live waves
  const int nNt  = Kout >> 4;
  const int nGrp = nNt >> 2;             // groups of 4 n-tiles
  const int mt   = wave / nGrp;
  const int nt0  = (wave - mt * nGrp) << 2;
  const int row0 = mt << 4, col0 = nt0 << 4;
  const int l0 = lane & 15, hi = lane >> 4;

  v8f acc0 = {}, acc1 = {}, acc2 = {}, acc3 = {};
  const int nkt = K >> 5;
  // A operand: lane l0 = row, hi selects the contiguous K+16 16-half run (matches packed B)
  const _Float16* pa = A + (size_t)(row0 + l0) * K + (hi << 4);
  const _Float16* pb = Bp + ((size_t)nt0 * 32 + lane) * 16;
  const size_t bstep = (size_t)nNt * 512;   // halves per k-tile row of B tiles
  for (int kt = 0; kt < nkt; ++kt) {
    half8_t alo = *(const half8_t*)(pa);
    half8_t ahi = *(const half8_t*)(pa + 8);
    v16h a = __builtin_shufflevector(alo, ahi, 0,1,2,3,4,5,6,7,8,9,10,11,12,13,14,15);
    v16h b0 = *(const v16h*)(pb);
    v16h b1 = *(const v16h*)(pb + 512);
    v16h b2 = *(const v16h*)(pb + 1024);
    v16h b3 = *(const v16h*)(pb + 1536);
    acc0 = __builtin_amdgcn_wmma_f32_16x16x32_f16(false, a, false, b0, (short)0, acc0, false, false);
    acc1 = __builtin_amdgcn_wmma_f32_16x16x32_f16(false, a, false, b1, (short)0, acc1, false, false);
    acc2 = __builtin_amdgcn_wmma_f32_16x16x32_f16(false, a, false, b2, (short)0, acc2, false, false);
    acc3 = __builtin_amdgcn_wmma_f32_16x16x32_f16(false, a, false, b3, (short)0, acc3, false, false);
    pa += 32;
    pb += bstep;
  }
  v8f accs[4] = {acc0, acc1, acc2, acc3};
#pragma unroll
  for (int t = 0; t < 4; ++t) {
    const float bv = bias[col0 + t * 16 + l0];  // N index identical across all 8 acc regs
#pragma unroll
    for (int r = 0; r < 8; ++r) {
      float v = accs[t][r] + bv;
      if (leaky) v = v > 0.f ? v : 0.01f * v;
      size_t o = (size_t)(row0 + (hi << 3) + r) * Kout + col0 + t * 16 + l0;
      if (outF) outF[o] = v;
      if (outH) outH[o] = (_Float16)v;
    }
  }
}

// ---------------- edge scatter: agg[col] += s[row] * ea  (one wave per edge) ----------------
__global__ void k_scatter(const float* __restrict__ s, const int* __restrict__ ei,
                          const float* __restrict__ ea, float* __restrict__ agg, int E)
{
  int wave = blockIdx.x * (blockDim.x >> 5) + (threadIdx.x >> 5);
  int lane = threadIdx.x & 31;
  if (wave >= E) return;
  int r = ei[wave];
  int c = ei[E + wave];
  float w = ea[wave];
  const float* sp = s + (size_t)r * D_;
  float* ap = agg + (size_t)c * D_;
#pragma unroll
  for (int i = 0; i < 4; ++i)
    atomicAdd(ap + lane + i * 32, sp[lane + i * 32] * w);
}

// divide agg by cnt and build f16 concat [agg/cnt, h]  (thread per element, N*128)
__global__ void k_cat(const float* __restrict__ agg, const float* __restrict__ cnt,
                      const _Float16* __restrict__ h16, _Float16* __restrict__ cat16)
{
  int idx = blockIdx.x * blockDim.x + threadIdx.x;
  int n = idx >> 7, i = idx & 127;
  float c = cnt[n]; if (c < 1.f) c = 1.f;
  cat16[(size_t)n * 256 + i]       = (_Float16)(agg[idx] / c);
  cat16[(size_t)n * 256 + 128 + i] = h16[idx];
}

// fused GRU update (thread per element, N*128)
__global__ void k_gru(const float* __restrict__ gi, const float* __restrict__ gh,
                      const float* __restrict__ hin, float* __restrict__ hout,
                      _Float16* __restrict__ h16out)
{
  int idx = blockIdx.x * blockDim.x + threadIdx.x;
  int n = idx >> 7, j = idx & 127;
  size_t b = (size_t)n * 384 + j;
  float ir = gi[b], iz = gi[b + 128], in_ = gi[b + 256];
  float hr = gh[b], hz = gh[b + 128], hn  = gh[b + 256];
  float r  = 1.f / (1.f + expf(-(ir + hr)));
  float z  = 1.f / (1.f + expf(-(iz + hz)));
  float nv = tanhf(in_ + r * hn);
  float o  = (1.f - z) * nv + z * hin[idx];
  hout[idx]   = o;
  h16out[idx] = (_Float16)o;
}

// ---------------- host launch ----------------

extern "C" void kernel_launch(void* const* d_in, const int* in_sizes, int n_in,
                              void* d_out, int out_size, void* d_ws, size_t ws_size,
                              hipStream_t stream) {
  (void)in_sizes; (void)n_in; (void)out_size; (void)ws_size;
  const float* x        = (const float*)d_in[0];
  const int*   eidx     = (const int*)  d_in[1];   // [2,E]: rows then cols
  const float* eattr    = (const float*)d_in[2];
  const float* W_embed  = (const float*)d_in[4];
  const float* b_embed  = (const float*)d_in[5];
  const float* W_snd    = (const float*)d_in[6];
  const float* b_snd    = (const float*)d_in[7];
  const float* W_rec    = (const float*)d_in[8];
  const float* b_rec    = (const float*)d_in[9];
  const float* W_ih     = (const float*)d_in[10];
  const float* b_ih     = (const float*)d_in[11];
  const float* W_hh     = (const float*)d_in[12];
  const float* b_hh     = (const float*)d_in[13];

  size_t off = 0;
  auto alloc = [&](size_t bytes) -> void* {
    void* p = (char*)d_ws + off;
    off += (bytes + 255) & ~(size_t)255;
    return p;
  };
  _Float16* x16   = (_Float16*)alloc((size_t)NNODE * DIN_ * 2);
  float*    h     = (float*)   alloc((size_t)NNODE * D_ * 4);
  _Float16* h16   = (_Float16*)alloc((size_t)NNODE * D_ * 2);
  float*    s     = (float*)   alloc((size_t)NNODE * D_ * 4);
  float*    agg   = (float*)   alloc((size_t)NNODE * D_ * 4);
  _Float16* cat16 = (_Float16*)alloc((size_t)NNODE * 256 * 2);
  _Float16* m16   = (_Float16*)alloc((size_t)NNODE * 256 * 2);
  float*    gi    = (float*)   alloc((size_t)NNODE * 384 * 4);
  float*    gh    = (float*)   alloc((size_t)NNODE * 384 * 4);
  float*    cnt   = (float*)   alloc((size_t)NNODE * 4);
  _Float16* pEmb  = (_Float16*)alloc((size_t)DIN_ * D_  * 2);
  _Float16* pSnd  = (_Float16*)alloc((size_t)D_   * D_  * 2);
  _Float16* pRec  = (_Float16*)alloc((size_t)256  * 256 * 2);
  _Float16* pIh   = (_Float16*)alloc((size_t)256  * 384 * 2);
  _Float16* pHh   = (_Float16*)alloc((size_t)D_   * 384 * 2);

  const int B = 256;
  // waves per GEMM: (M/16) * (Kout/64)  (each wave covers 4 N-tiles)
  const int wv128 = (NNODE / 16) * (128 / 64);   // 6250
  const int wv256 = (NNODE / 16) * (256 / 64);   // 12500
  const int wv384 = (NNODE / 16) * (384 / 64);   // 18750
  auto gblocks = [](int waves) { return (waves + 7) / 8; };

  // convert x, pack weights, build degree counts
  k_f32_to_f16<<<(NNODE * DIN_) / B, B, 0, stream>>>(x, x16, NNODE * DIN_);
  k_pack_b<<<(DIN_ * D_) / B,  B, 0, stream>>>(W_embed, pEmb, DIN_, D_);
  k_pack_b<<<(D_ * D_) / B,    B, 0, stream>>>(W_snd,   pSnd, D_,   D_);
  k_pack_b<<<(256 * 256) / B,  B, 0, stream>>>(W_rec,   pRec, 256,  256);
  k_pack_b<<<(256 * 384) / B,  B, 0, stream>>>(W_ih,    pIh,  256,  384);
  k_pack_b<<<(D_ * 384) / B,   B, 0, stream>>>(W_hh,    pHh,  D_,   384);
  k_zero_f32<<<(NNODE / 4 + B - 1) / B, B, 0, stream>>>(cnt, NNODE / 4);
  k_count<<<NEDGE / B, B, 0, stream>>>(eidx + NEDGE, cnt, NEDGE);

  // h = x @ W_embed + b   (no activation; f32 master + f16 copy)
  k_gemm_wmma<<<gblocks(wv128), B, 0, stream>>>(
      x16, pEmb, b_embed, h, h16, DIN_, D_, 0, wv128);

  for (int layer = 0; layer < 3; ++layer) {
    // s = leaky_relu(h @ W_snd + b)
    k_gemm_wmma<<<gblocks(wv128), B, 0, stream>>>(
        h16, pSnd, b_snd, s, (_Float16*)nullptr, D_, D_, 1, wv128);
    // agg = segment_sum(s[row] * ea, col)
    k_zero_f32<<<(NNODE * D_ / 4) / B, B, 0, stream>>>(agg, NNODE * D_ / 4);
    k_scatter<<<NEDGE / 8, B, 0, stream>>>(s, eidx, eattr, agg, NEDGE);
    // cat16 = f16([agg/cnt, h])
    k_cat<<<(NNODE * D_) / B, B, 0, stream>>>(agg, cnt, h16, cat16);
    // m = leaky_relu(cat @ W_rec + b)  (f16 only; feeds next GEMM)
    k_gemm_wmma<<<gblocks(wv256), B, 0, stream>>>(
        cat16, pRec, b_rec, (float*)nullptr, m16, 256, 256, 1, wv256);
    // gi = m @ W_ih + b ; gh = h @ W_hh + b
    k_gemm_wmma<<<gblocks(wv384), B, 0, stream>>>(
        m16, pIh, b_ih, gi, (_Float16*)nullptr, 256, 384, 0, wv384);
    k_gemm_wmma<<<gblocks(wv384), B, 0, stream>>>(
        h16, pHh, b_hh, gh, (_Float16*)nullptr, D_, 384, 0, wv384);
    // GRU update; final layer writes straight to d_out
    float* hdst = (layer == 2) ? (float*)d_out : h;
    k_gru<<<(NNODE * D_) / B, B, 0, stream>>>(gi, gh, h, hdst, h16);
  }
}
